// GNN_81372450390724
// MI455X (gfx1250) — matrix operand order repaired and checked by reading, not compiled, not run on hardware
//
#include <hip/hip_runtime.h>
#include <hip/hip_bf16.h>

typedef _Float16 f16;
typedef __attribute__((ext_vector_type(16))) _Float16 v16h;
typedef __attribute__((ext_vector_type(8)))  _Float16 v8h;
typedef __attribute__((ext_vector_type(4)))  _Float16 v4h;
typedef __attribute__((ext_vector_type(8)))  float    v8f;

#define NNODE  50000
#define MPAD   50048    // padded to multiple of 64 for guard-free 2x2 WMMA tiling
#define FDIM   256
#define NEDGE  200000
#define NHEAD  4
#define CDIM   128
#define HC     512      // NHEAD * CDIM
#define NGRAPH 256
#define NCLS   10

// ---------- helpers ----------
__device__ __forceinline__ unsigned f2ord(float f) {
    unsigned u = __float_as_uint(f);
    return (u & 0x80000000u) ? ~u : (u | 0x80000000u);
}
__device__ __forceinline__ float ord2f(unsigned u) {
    unsigned b = (u & 0x80000000u) ? (u ^ 0x80000000u) : ~u;
    return __uint_as_float(b);
}
__device__ __forceinline__ float elu1(float x) {
    return x > 0.0f ? x : expm1f(x);
}

// ---------- utility kernels ----------
__global__ void zero_u32(unsigned* __restrict__ p, int n) {
    int i = blockIdx.x * blockDim.x + threadIdx.x;
    if (i < n) p[i] = 0u;
}

__global__ void cvt_f32_f16(const float* __restrict__ s, f16* __restrict__ d, int n) {
    int i = blockIdx.x * blockDim.x + threadIdx.x;
    if (i < n) d[i] = (f16)s[i];
}

// W is [K, Nc] row-major fp32 -> Wt is [Nc, K] row-major f16 (column-major of W)
__global__ void transpose_w(const float* __restrict__ W, f16* __restrict__ Wt,
                            int K, int Nc) {
    int i = blockIdx.x * blockDim.x + threadIdx.x;
    if (i < K * Nc) {
        int k = i / Nc, n = i % Nc;
        Wt[(size_t)n * K + k] = (f16)W[i];
    }
}

// ---------- WMMA GEMM:  C[Mpad,Nc] = A[Mpad,K] @ Bt^T + bias ----------
// A:  [Mpad, K] row-major f16 (rows >= NNODE are zero-padded)
// Bt: [Nc, K]   row-major f16 (pre-transposed weight)
// Each wave computes a 32x32 tile as 2x2 WMMA fragments (4 accumulators):
// per 32-k step: 2 A-frag loads + 2 B-frag loads feed 4 v_wmma -> 2x operand reuse.
// Block = 8 waves (2 Mtiles x 4 Ntiles) = 64 rows x 128 cols.
// Fragment layouts used here (wave32; A/B contiguous k is this code's choice, not the ISA table order):
//   A 16x32 f16 : lane l holds row m=l&15; k-halves (l>>4)*16+[0..15] (contiguous)
//   B 32x16 f16 : lane l holds col n=l&15; k-halves (l>>4)*16+[0..15] (contiguous)
//   C/D 16x16 f32: lane l col n=l&15; VGPR v row m = v + 8*(l>>4)
__global__ __launch_bounds__(256)
void wmma_gemm(const f16* __restrict__ A, const f16* __restrict__ Bt,
               const float* __restrict__ bias,
               f16* __restrict__ Ch, float* __restrict__ Cf,
               int K, int Nc) {
    const int lane = threadIdx.x & 31;
    const int wave = threadIdx.x >> 5;
    const int mw = wave >> 2;       // 0..1
    const int nw = wave & 3;        // 0..3
    const int m0 = (blockIdx.x << 6) + (mw << 5);
    const int n0 = (blockIdx.y << 7) + (nw << 5);
    const int r  = lane & 15;
    const int hi = lane >> 4;

    const f16* Ap0 = A  + (size_t)(m0 + r) * K + (hi << 4);
    const f16* Ap1 = Ap0 + (size_t)16 * K;
    const f16* Bp0 = Bt + (size_t)(n0 + r) * K + (hi << 4);
    const f16* Bp1 = Bp0 + (size_t)16 * K;

    v8f acc00 = {}, acc01 = {}, acc10 = {}, acc11 = {};
    for (int kk = 0; kk < K; kk += 32) {
        v8h a0lo = *(const v8h*)(Ap0 + kk);
        v8h a0hi = *(const v8h*)(Ap0 + kk + 8);
        v16h A0 = __builtin_shufflevector(a0lo, a0hi,
                      0,1,2,3,4,5,6,7, 8,9,10,11,12,13,14,15);
        v8h a1lo = *(const v8h*)(Ap1 + kk);
        v8h a1hi = *(const v8h*)(Ap1 + kk + 8);
        v16h A1 = __builtin_shufflevector(a1lo, a1hi,
                      0,1,2,3,4,5,6,7, 8,9,10,11,12,13,14,15);
        v16h B0 = *(const v16h*)(Bp0 + kk);
        v16h B1 = *(const v16h*)(Bp1 + kk);
        acc00 = __builtin_amdgcn_wmma_f32_16x16x32_f16(false, A0, false, B0,
                                                       (short)0, acc00, false, false);
        acc01 = __builtin_amdgcn_wmma_f32_16x16x32_f16(false, A0, false, B1,
                                                       (short)0, acc01, false, false);
        acc10 = __builtin_amdgcn_wmma_f32_16x16x32_f16(false, A1, false, B0,
                                                       (short)0, acc10, false, false);
        acc11 = __builtin_amdgcn_wmma_f32_16x16x32_f16(false, A1, false, B1,
                                                       (short)0, acc11, false, false);
    }

    const int nA = n0 + r;
    const int nB = nA + 16;
    const float bnA = bias[nA];
    const float bnB = bias[nB];
    const int mb0 = m0 + (hi << 3);
    const int mb1 = mb0 + 16;
    if (Ch) {
#pragma unroll
        for (int v = 0; v < 8; ++v) {
            Ch[(size_t)(mb0 + v) * Nc + nA] = (f16)(acc00[v] + bnA);
            Ch[(size_t)(mb0 + v) * Nc + nB] = (f16)(acc01[v] + bnB);
            Ch[(size_t)(mb1 + v) * Nc + nA] = (f16)(acc10[v] + bnA);
            Ch[(size_t)(mb1 + v) * Nc + nB] = (f16)(acc11[v] + bnB);
        }
    } else {
#pragma unroll
        for (int v = 0; v < 8; ++v) {
            Cf[(size_t)(mb0 + v) * Nc + nA] = acc00[v] + bnA;
            Cf[(size_t)(mb0 + v) * Nc + nB] = acc01[v] + bnB;
            Cf[(size_t)(mb1 + v) * Nc + nA] = acc10[v] + bnA;
            Cf[(size_t)(mb1 + v) * Nc + nB] = acc11[v] + bnB;
        }
    }
}

// ---------- attention: pass 1, per-(edge,head) score + segment max ----------
__global__ __launch_bounds__(256)
void attn_score(const f16* __restrict__ q, const f16* __restrict__ k,
                const int* __restrict__ src, const int* __restrict__ dst,
                float* __restrict__ alpha, unsigned* __restrict__ mseg, int EH) {
    int wid = (blockIdx.x * blockDim.x + threadIdx.x) >> 5;
    if (wid >= EH) return;
    int lane = threadIdx.x & 31;
    int e = wid >> 2, h = wid & 3;
    int sj = src[e], di = dst[e];
    v4h qv = ((const v4h*)(q + (size_t)di * HC + h * CDIM))[lane];
    v4h kv = ((const v4h*)(k + (size_t)sj * HC + h * CDIM))[lane];
    float acc = 0.0f;
#pragma unroll
    for (int i = 0; i < 4; ++i) acc += (float)qv[i] * (float)kv[i];
    for (int o = 16; o > 0; o >>= 1) acc += __shfl_xor(acc, o, 32);
    if (lane == 0) {
        float a = acc * 0.08838834764831845f;  // 1/sqrt(128)
        alpha[wid] = a;
        atomicMax(mseg + (size_t)di * NHEAD + h, f2ord(a));
    }
}

// ---------- attention: pass 2, exp(alpha-m), segment sum ----------
__global__ void attn_exp(float* __restrict__ alpha, const unsigned* __restrict__ mseg,
                         const int* __restrict__ dst, float* __restrict__ sseg, int EH) {
    int gid = blockIdx.x * blockDim.x + threadIdx.x;
    if (gid >= EH) return;
    int e = gid >> 2, h = gid & 3;
    int di = dst[e];
    float mm = ord2f(mseg[(size_t)di * NHEAD + h]);
    float ex = expf(alpha[gid] - mm);
    alpha[gid] = ex;
    atomicAdd(sseg + (size_t)di * NHEAD + h, ex);
}

// ---------- attention: pass 3, weighted message scatter ----------
__global__ __launch_bounds__(256)
void attn_agg(const float* __restrict__ alpha, const float* __restrict__ sseg,
              const f16* __restrict__ v, const int* __restrict__ src,
              const int* __restrict__ dst, float* __restrict__ msg, int EH) {
    int wid = (blockIdx.x * blockDim.x + threadIdx.x) >> 5;
    if (wid >= EH) return;
    int lane = threadIdx.x & 31;
    int e = wid >> 2, h = wid & 3;
    int sj = src[e], di = dst[e];
    float a = alpha[wid] / (sseg[(size_t)di * NHEAD + h] + 1e-16f);
    v4h vv = ((const v4h*)(v + (size_t)sj * HC + h * CDIM))[lane];
    float* mp = msg + (size_t)di * HC + h * CDIM + lane * 4;
#pragma unroll
    for (int i = 0; i < 4; ++i) atomicAdd(mp + i, a * (float)vv[i]);
}

// ---------- layer finalizers ----------
__global__ void finalize1(const float* __restrict__ msg, const float* __restrict__ skip,
                          f16* __restrict__ h1, int n) {
    int i = blockIdx.x * blockDim.x + threadIdx.x;
    if (i < n) h1[i] = (f16)elu1(msg[i] + skip[i]);
}

__global__ void finalize2(const float* __restrict__ msg, const float* __restrict__ skip,
                          float* __restrict__ h2, int n) {
    int i = blockIdx.x * blockDim.x + threadIdx.x;
    if (i >= n) return;
    int nn = i >> 7, c = i & 127;
    size_t b = (size_t)nn * HC + c;
    float m = 0.25f * (msg[b] + msg[b + 128] + msg[b + 256] + msg[b + 384]);
    h2[i] = elu1(m + skip[i]);
}

// ---------- pooling ----------
__global__ void pool_add(const float* __restrict__ h2, const int* __restrict__ batch,
                         float* __restrict__ pool, int n) {
    int i = blockIdx.x * blockDim.x + threadIdx.x;
    if (i < n) atomicAdd(pool + (size_t)batch[i >> 7] * CDIM + (i & 127), h2[i]);
}
__global__ void cnt_add(const int* __restrict__ batch, float* __restrict__ cnt, int n) {
    int i = blockIdx.x * blockDim.x + threadIdx.x;
    if (i < n) atomicAdd(cnt + batch[i], 1.0f);
}

// ---------- classifier head + log_softmax (one wave per graph) ----------
__global__ void head_kernel(const float* __restrict__ pool, const float* __restrict__ cnt,
                            const float* __restrict__ Wfc, const float* __restrict__ bfc,
                            float* __restrict__ out) {
    int g = blockIdx.x;
    int j = threadIdx.x;
    float inv = 1.0f / fmaxf(cnt[g], 1.0f);
    float logit = -3.402823466e38f;
    if (j < NCLS) {
        float acc = bfc[j];
        for (int c = 0; c < CDIM; ++c)
            acc += (pool[(size_t)g * CDIM + c] * inv) * Wfc[c * NCLS + j];
        logit = acc;
    }
    float mx = logit;
    for (int o = 16; o > 0; o >>= 1) mx = fmaxf(mx, __shfl_xor(mx, o, 32));
    float ex = (j < NCLS) ? expf(logit - mx) : 0.0f;
    float se = ex;
    for (int o = 16; o > 0; o >>= 1) se += __shfl_xor(se, o, 32);
    if (j < NCLS) out[(size_t)g * NCLS + j] = logit - mx - logf(se);
}

// ---------- host orchestration ----------
extern "C" void kernel_launch(void* const* d_in, const int* in_sizes, int n_in,
                              void* d_out, int out_size, void* d_ws, size_t ws_size,
                              hipStream_t stream) {
    (void)in_sizes; (void)n_in; (void)out_size; (void)ws_size;
    const float* x    = (const float*)d_in[0];
    const int*   ei   = (const int*)d_in[1];
    const int*   batch= (const int*)d_in[2];
    const float* Wq1 = (const float*)d_in[3];  const float* bq1 = (const float*)d_in[4];
    const float* Wk1 = (const float*)d_in[5];  const float* bk1 = (const float*)d_in[6];
    const float* Wv1 = (const float*)d_in[7];  const float* bv1 = (const float*)d_in[8];
    const float* Ws1 = (const float*)d_in[9];  const float* bs1 = (const float*)d_in[10];
    const float* Wq2 = (const float*)d_in[11]; const float* bq2 = (const float*)d_in[12];
    const float* Wk2 = (const float*)d_in[13]; const float* bk2 = (const float*)d_in[14];
    const float* Wv2 = (const float*)d_in[15]; const float* bv2 = (const float*)d_in[16];
    const float* Ws2 = (const float*)d_in[17]; const float* bs2 = (const float*)d_in[18];
    const float* Wfc = (const float*)d_in[19]; const float* bfc = (const float*)d_in[20];
    const int* src = ei;
    const int* dst = ei + NEDGE;

    char* ws = (char*)d_ws;
    size_t off = 0;
    auto carve = [&](size_t bytes) -> char* {
        char* p = ws + off;
        off += (bytes + 255) & ~(size_t)255;
        return p;
    };

    f16* xh  = (f16*)carve((size_t)MPAD * FDIM * 2);
    f16* w1q = (f16*)carve((size_t)FDIM * HC * 2);
    f16* w1k = (f16*)carve((size_t)FDIM * HC * 2);
    f16* w1v = (f16*)carve((size_t)FDIM * HC * 2);
    f16* w1s = (f16*)carve((size_t)FDIM * HC * 2);
    f16* w2q = (f16*)carve((size_t)HC * HC * 2);
    f16* w2k = (f16*)carve((size_t)HC * HC * 2);
    f16* w2v = (f16*)carve((size_t)HC * HC * 2);
    f16* w2s = (f16*)carve((size_t)HC * CDIM * 2);
    f16* qh  = (f16*)carve((size_t)MPAD * HC * 2);
    f16* kh  = (f16*)carve((size_t)MPAD * HC * 2);
    f16* vh  = (f16*)carve((size_t)MPAD * HC * 2);
    float*    skip  = (float*)carve((size_t)MPAD * HC * 4);
    float*    alpha = (float*)carve((size_t)NEDGE * NHEAD * 4);
    unsigned* mseg  = (unsigned*)carve((size_t)NNODE * NHEAD * 4);
    float*    sseg  = (float*)carve((size_t)NNODE * NHEAD * 4);
    float*    msg   = (float*)carve((size_t)NNODE * HC * 4);
    f16*      h1h   = (f16*)carve((size_t)MPAD * HC * 2);
    float*    h2    = (float*)carve((size_t)NNODE * CDIM * 4);
    float*    pool  = (float*)carve((size_t)NGRAPH * CDIM * 4);
    float*    cnt   = (float*)carve((size_t)NGRAPH * 4);

    const int EH = NEDGE * NHEAD;
    const int T = 256;
    auto blks = [](int n, int t) { return (n + t - 1) / t; };
    const int PADF  = (MPAD - NNODE) * FDIM / 2;  // pad words (u32) of xh
    const int PADHC = (MPAD - NNODE) * HC / 2;    // pad words (u32) of h1h

    // convert x + zero A-input pad rows + transpose weights
    cvt_f32_f16<<<blks(NNODE * FDIM, T), T, 0, stream>>>(x, xh, NNODE * FDIM);
    zero_u32<<<blks(PADF, T), T, 0, stream>>>((unsigned*)(xh + (size_t)NNODE * FDIM), PADF);
    transpose_w<<<blks(FDIM * HC, T), T, 0, stream>>>(Wq1, w1q, FDIM, HC);
    transpose_w<<<blks(FDIM * HC, T), T, 0, stream>>>(Wk1, w1k, FDIM, HC);
    transpose_w<<<blks(FDIM * HC, T), T, 0, stream>>>(Wv1, w1v, FDIM, HC);
    transpose_w<<<blks(FDIM * HC, T), T, 0, stream>>>(Ws1, w1s, FDIM, HC);
    transpose_w<<<blks(HC * HC, T), T, 0, stream>>>(Wq2, w2q, HC, HC);
    transpose_w<<<blks(HC * HC, T), T, 0, stream>>>(Wk2, w2k, HC, HC);
    transpose_w<<<blks(HC * HC, T), T, 0, stream>>>(Wv2, w2v, HC, HC);
    transpose_w<<<blks(HC * CDIM, T), T, 0, stream>>>(Ws2, w2s, HC, CDIM);

    // ---- layer 1 GEMMs ----
    dim3 g1(MPAD / 64, HC / 128);
    wmma_gemm<<<g1, T, 0, stream>>>(xh, w1q, bq1, qh, nullptr, FDIM, HC);
    wmma_gemm<<<g1, T, 0, stream>>>(xh, w1k, bk1, kh, nullptr, FDIM, HC);
    wmma_gemm<<<g1, T, 0, stream>>>(xh, w1v, bv1, vh, nullptr, FDIM, HC);
    wmma_gemm<<<g1, T, 0, stream>>>(xh, w1s, bs1, nullptr, skip, FDIM, HC);

    // ---- layer 1 attention ----
    zero_u32<<<blks(NNODE * NHEAD, T), T, 0, stream>>>(mseg, NNODE * NHEAD);
    zero_u32<<<blks(NNODE * NHEAD, T), T, 0, stream>>>((unsigned*)sseg, NNODE * NHEAD);
    zero_u32<<<blks(NNODE * HC, T), T, 0, stream>>>((unsigned*)msg, NNODE * HC);
    attn_score<<<blks(EH * 32, T), T, 0, stream>>>(qh, kh, src, dst, alpha, mseg, EH);
    attn_exp<<<blks(EH, T), T, 0, stream>>>(alpha, mseg, dst, sseg, EH);
    attn_agg<<<blks(EH * 32, T), T, 0, stream>>>(alpha, sseg, vh, src, dst, msg, EH);
    finalize1<<<blks(NNODE * HC, T), T, 0, stream>>>(msg, skip, h1h, NNODE * HC);
    zero_u32<<<blks(PADHC, T), T, 0, stream>>>((unsigned*)(h1h + (size_t)NNODE * HC), PADHC);

    // ---- layer 2 GEMMs (reuse q/k/v/skip buffers) ----
    dim3 g2(MPAD / 64, HC / 128);
    wmma_gemm<<<g2, T, 0, stream>>>(h1h, w2q, bq2, qh, nullptr, HC, HC);
    wmma_gemm<<<g2, T, 0, stream>>>(h1h, w2k, bk2, kh, nullptr, HC, HC);
    wmma_gemm<<<g2, T, 0, stream>>>(h1h, w2v, bv2, vh, nullptr, HC, HC);
    dim3 g2s(MPAD / 64, CDIM / 128);
    wmma_gemm<<<g2s, T, 0, stream>>>(h1h, w2s, bs2, nullptr, skip, HC, CDIM);

    // ---- layer 2 attention ----
    zero_u32<<<blks(NNODE * NHEAD, T), T, 0, stream>>>(mseg, NNODE * NHEAD);
    zero_u32<<<blks(NNODE * NHEAD, T), T, 0, stream>>>((unsigned*)sseg, NNODE * NHEAD);
    zero_u32<<<blks(NNODE * HC, T), T, 0, stream>>>((unsigned*)msg, NNODE * HC);
    attn_score<<<blks(EH * 32, T), T, 0, stream>>>(qh, kh, src, dst, alpha, mseg, EH);
    attn_exp<<<blks(EH, T), T, 0, stream>>>(alpha, mseg, dst, sseg, EH);
    attn_agg<<<blks(EH * 32, T), T, 0, stream>>>(alpha, sseg, vh, src, dst, msg, EH);
    finalize2<<<blks(NNODE * CDIM, T), T, 0, stream>>>(msg, skip, h2, NNODE * CDIM);

    // ---- pooling + head ----
    zero_u32<<<blks(NGRAPH * CDIM, T), T, 0, stream>>>((unsigned*)pool, NGRAPH * CDIM);
    zero_u32<<<blks(NGRAPH, T), T, 0, stream>>>((unsigned*)cnt, NGRAPH);
    pool_add<<<blks(NNODE * CDIM, T), T, 0, stream>>>(h2, batch, pool, NNODE * CDIM);
    cnt_add<<<blks(NNODE, T), T, 0, stream>>>(batch, cnt, NNODE);
    head_kernel<<<NGRAPH, 32, 0, stream>>>(pool, cnt, Wfc, bfc, (float*)d_out);
}
